// ClientGCN_81415400063394
// MI455X (gfx1250) — matrix operand order repaired and checked by reading, not compiled, not run on hardware
//
#include <hip/hip_runtime.h>

// ---------------------------------------------------------------------------
// GCN 2-layer forward for MI455X (gfx1250, wave32, WMMA).
// N=100000 nodes, E=1000000 edges, IN=128, H=64.
// Edge scatter dominates (~0.5 GB random traffic/layer vs 23.3 TB/s HBM and a
// 192 MB L2 that holds every intermediate); GEMMs run on
// v_wmma_f32_16x16x32_bf16 with pre-packed operand fragments so the matrix
// math is pure b128 loads + wmma.
// ---------------------------------------------------------------------------

#define GCN_N   100000
#define GCN_E   1000000
#define GCN_IN  128
#define GCN_H   64
#define GCN_EPS 1e-5f

typedef __attribute__((ext_vector_type(16))) __bf16 v16bf;
typedef __attribute__((ext_vector_type(8)))  __bf16 v8bf;
typedef __attribute__((ext_vector_type(8)))  float  v8f;

__device__ __forceinline__ __bf16 to_bf16(float f) {
    union { float f; unsigned u; } v; v.f = f;
    unsigned r = v.u + 0x7FFFu + ((v.u >> 16) & 1u);   // round-to-nearest-even
    unsigned short h = (unsigned short)(r >> 16);
    return __builtin_bit_cast(__bf16, h);
}

// ---------------- elementwise helpers ----------------

__global__ void gcn_f32_to_bf16(const float* __restrict__ in,
                                __bf16* __restrict__ out, long n) {
    long i = (long)blockIdx.x * blockDim.x + threadIdx.x;
    if (i < n) out[i] = to_bf16(in[i]);
}

__global__ void gcn_fill1(float* __restrict__ p, int n) {
    int i = blockIdx.x * blockDim.x + threadIdx.x;
    if (i < n) p[i] = 1.0f;                 // self-loop contributes 1 to degree
}

__global__ void gcn_deg_edges(const int* __restrict__ dst,
                              float* __restrict__ deg, int e) {
    int i = blockIdx.x * blockDim.x + threadIdx.x;
    if (i < e) atomicAdd(&deg[dst[i]], 1.0f);
}

__global__ void gcn_rsqrt_inplace(float* __restrict__ p, int n) {
    int i = blockIdx.x * blockDim.x + threadIdx.x;
    if (i < n) p[i] = rsqrtf(p[i]);         // deg >= 1 always (self loop)
}

__global__ void gcn_zero_stats(float* __restrict__ stats) {
    if (threadIdx.x < 256) stats[threadIdx.x] = 0.0f;
}

// ---------------- weight pre-pack into WMMA B-fragment layout ----------------
// B-frag for v_wmma_f32_16x16x32_bf16 (32x16 bf16): lanes 0-15 hold col N=lane,
// K=k0+{0..7,16..23}; lanes 16-31 hold K=k0+{8..15,24..31}. Pack W[K,64] (f32, row-major) as
//   Bp[tile][lane][j]  with tile = (k0/32)*4 + (n0/16), j = 0..15
// so each lane's whole fragment is 32 contiguous, 32B-aligned bytes.
__global__ void gcn_pack_B(const float* __restrict__ W,
                           __bf16* __restrict__ Bp, int K) {
    int idx = blockIdx.x * blockDim.x + threadIdx.x;   // tile*512 + lane*16 + j
    int total = (K >> 5) * 4 * 512;
    if (idx >= total) return;
    int j    = idx & 15;
    int lane = (idx >> 4) & 31;
    int tile = idx >> 9;
    int n0 = (tile & 3) * 16;
    int k0 = (tile >> 2) * 32;
    int k = k0 + (lane >> 4) * 8 + (j & 7) + (j >> 3) * 16;
    int n = n0 + (lane & 15);
    Bp[idx] = to_bf16(W[(size_t)k * GCN_H + n]);
}

// ---------------- WMMA GEMM: C[M,64] = A[M,K](bf16) @ W (pre-packed) --------
// One wave -> one 16x16 C tile. Block = 256 threads = 8 waves = 32 rows x 64 cols.
// A-frag (16-bit 16x32): lanes 0-15 row M=lane%16, K = k0+{0..7,16..23};
//                        lanes 16-31 same rows, K offset +8.
// C/D (f32 16x16): VGPR r -> row r + 8*(lane>=16), col = lane%16.
template <int KDIM>
__global__ void gcn_gemm_bf16(const __bf16* __restrict__ A,
                              const __bf16* __restrict__ Bp,
                              float* __restrict__ C, int M) {
    const int wave = threadIdx.x >> 5;
    const int lane = threadIdx.x & 31;
    const int m0 = blockIdx.x * 32 + (wave >> 2) * 16;   // M tile
    const int n0 = (wave & 3) * 16;                      // N tile (H=64 -> 4 tiles)
    if (m0 >= M) return;                                 // wave-uniform guard
    const int lmod = lane & 15;
    const int lhi  = lane >> 4;                          // 0 | 1
    const int m = m0 + lmod;

    v8f acc = {};
#pragma unroll
    for (int k0 = 0; k0 < KDIM; k0 += 32) {
        const __bf16* arow = A + (size_t)m * KDIM + k0 + lhi * 8;
        v8bf alo = *(const v8bf*)(arow);        // K = k0+8*lhi    .. +7   (b128)
        v8bf ahi = *(const v8bf*)(arow + 16);   // K = k0+8*lhi+16 .. +23  (b128)
        v16bf a;
#pragma unroll
        for (int i = 0; i < 8; ++i) { a[i] = alo[i]; a[8 + i] = ahi[i]; }

        // whole B fragment: one 32B contiguous read (2 x b128)
        const int tile = (k0 >> 5) * 4 + (n0 >> 4);
        v16bf b = *(const v16bf*)(Bp + (size_t)tile * 512 + lane * 16);

        acc = __builtin_amdgcn_wmma_f32_16x16x32_bf16(
                  false, a, false, b, (short)0, acc, false, false);
    }
#pragma unroll
    for (int r = 0; r < 8; ++r)
        C[(size_t)(m0 + r + lhi * 8) * GCN_H + (n0 + lmod)] = acc[r];
}

// ---------------- aggregation ----------------

// agg[i,f] = h[i,f] * dinv[i]^2   (self-loop term; also zero-initializes agg)
__global__ void gcn_selfloop(const float* __restrict__ h,
                             const float* __restrict__ dinv,
                             float* __restrict__ agg, long n64) {
    long i = (long)blockIdx.x * blockDim.x + threadIdx.x;
    if (i < n64) {
        float di = dinv[i >> 6];
        agg[i] = h[i] * di * di;
    }
}

// agg[d,f] += h[s,f] * dinv[s]*dinv[d]   -- 4 edges / 256-thread block
__global__ void gcn_scatter(const int* __restrict__ src,
                            const int* __restrict__ dst,
                            const float* __restrict__ dinv,
                            const float* __restrict__ h,
                            float* __restrict__ agg, int e) {
    int edge = blockIdx.x * 4 + (threadIdx.x >> 6);
    int f = threadIdx.x & 63;
    if (edge >= e) return;
    int s = src[edge], d = dst[edge];
    const float* hrow = h + (size_t)s * GCN_H;
    __builtin_prefetch(hrow, 0, 0);                 // global_prefetch_b8
    float w = dinv[s] * dinv[d];
    atomicAdd(&agg[(size_t)d * GCN_H + f], hrow[f] * w);
}

// ---------------- batch norm (training-mode batch stats) ----------------

// per-feature sum / sumsq of (agg + bias), 1024 rows per block, LDS reduce
__global__ void gcn_bn_stats(const float* __restrict__ agg,
                             const float* __restrict__ bias,
                             float* __restrict__ stats, int n) {
    __shared__ float ls[256];
    __shared__ float lq[256];
    int f = threadIdx.x & 63;
    int slot = threadIdx.x >> 6;                    // 0..3
    int r0 = blockIdx.x * 1024;
    int rend = r0 + 1024; if (rend > n) rend = n;
    float bf = bias[f];
    float s = 0.0f, q = 0.0f;
    for (int r = r0 + slot; r < rend; r += 4) {
        float y = agg[(size_t)r * GCN_H + f] + bf;
        s += y; q += y * y;
    }
    ls[threadIdx.x] = s; lq[threadIdx.x] = q;
    __syncthreads();
    if (threadIdx.x < 64) {
        float ts = ls[f] + ls[64 + f] + ls[128 + f] + ls[192 + f];
        float tq = lq[f] + lq[64 + f] + lq[128 + f] + lq[192 + f];
        atomicAdd(&stats[f], ts);
        atomicAdd(&stats[64 + f], tq);
    }
}

// stats[128+f] = scale, stats[192+f] = shift   (y*scale + shift, then ReLU)
__global__ void gcn_bn_finalize(const float* __restrict__ g,
                                const float* __restrict__ bt,
                                float* __restrict__ stats, float invN) {
    int f = threadIdx.x;
    if (f < 64) {
        float mean = stats[f] * invN;
        float var  = stats[64 + f] * invN - mean * mean;   // biased var
        float sc = g[f] * rsqrtf(var + GCN_EPS);
        stats[128 + f] = sc;
        stats[192 + f] = bt[f] - mean * sc;
    }
}

template <bool BF16OUT>
__global__ void gcn_bn_apply(const float* __restrict__ agg,
                             const float* __restrict__ bias,
                             const float* __restrict__ stats,
                             float* __restrict__ outf,
                             __bf16* __restrict__ outb, long n64) {
    long i = (long)blockIdx.x * blockDim.x + threadIdx.x;
    if (i < n64) {
        int f = (int)(i & 63);
        float y = agg[i] + bias[f];
        float z = fmaxf(y * stats[128 + f] + stats[192 + f], 0.0f);
        if (BF16OUT) outb[i] = to_bf16(z);
        else         outf[i] = z;
    }
}

// ---------------------------------------------------------------------------

extern "C" void kernel_launch(void* const* d_in, const int* in_sizes, int n_in,
                              void* d_out, int out_size, void* d_ws, size_t ws_size,
                              hipStream_t stream) {
    (void)in_sizes; (void)n_in; (void)out_size; (void)ws_size;
    const float* x   = (const float*)d_in[0];
    const int*   ei  = (const int*)d_in[1];      // [2,E]: row0 = src, row1 = dst
    const float* W1  = (const float*)d_in[2];
    const float* b1  = (const float*)d_in[3];
    const float* g1  = (const float*)d_in[4];
    const float* bt1 = (const float*)d_in[5];
    const float* W2  = (const float*)d_in[6];
    const float* b2  = (const float*)d_in[7];
    const float* g2  = (const float*)d_in[8];
    const float* bt2 = (const float*)d_in[9];
    float* out = (float*)d_out;

    const int* src = ei;
    const int* dst = ei + GCN_E;

    // -------- workspace carve-up (256B aligned) --------
    size_t off = 0;
    auto carve = [&](size_t bytes) {
        size_t o = off;
        off = (off + bytes + 255) & ~(size_t)255;
        return o;
    };
    char* ws = (char*)d_ws;
    float*  h     = (float*)(ws + carve((size_t)GCN_N * GCN_H * 4));   // GEMM out
    float*  agg   = (float*)(ws + carve((size_t)GCN_N * GCN_H * 4));   // aggregated
    __bf16* act   = (__bf16*)(ws + carve((size_t)GCN_N * GCN_H * 2));  // layer-1 act (A of layer 2)
    __bf16* xbf   = (__bf16*)(ws + carve((size_t)GCN_N * GCN_IN * 2)); // x in bf16
    float*  dinv  = (float*)(ws + carve((size_t)GCN_N * 4));           // deg -> rsqrt(deg)
    __bf16* W1p   = (__bf16*)(ws + carve((size_t)GCN_IN * GCN_H * 2)); // packed B frags
    __bf16* W2p   = (__bf16*)(ws + carve((size_t)GCN_H * GCN_H * 2));
    float*  stats = (float*)(ws + carve(256 * 4));

    const long NH  = (long)GCN_N * GCN_H;
    const long NIN = (long)GCN_N * GCN_IN;
    const int  T   = 256;
    auto blocks = [&](long n) { return (unsigned)((n + T - 1) / T); };

    // -------- degrees / symmetric normalization --------
    gcn_fill1<<<blocks(GCN_N), T, 0, stream>>>(dinv, GCN_N);
    gcn_deg_edges<<<blocks(GCN_E), T, 0, stream>>>(dst, dinv, GCN_E);
    gcn_rsqrt_inplace<<<blocks(GCN_N), T, 0, stream>>>(dinv, GCN_N);

    // -------- bf16 operand preparation --------
    gcn_f32_to_bf16<<<blocks(NIN), T, 0, stream>>>(x, xbf, NIN);
    gcn_pack_B<<<blocks((long)GCN_IN * GCN_H), T, 0, stream>>>(W1, W1p, GCN_IN);
    gcn_pack_B<<<blocks((long)GCN_H * GCN_H), T, 0, stream>>>(W2, W2p, GCN_H);

    const unsigned gemmGrid    = (GCN_N + 31) / 32;       // 32 rows / block (8 waves)
    const unsigned scatterGrid = (GCN_E + 3) / 4;         // 4 edges / block
    const unsigned statsGrid   = (GCN_N + 1023) / 1024;
    const float invN = 1.0f / (float)GCN_N;

    // ================= layer 1 =================
    gcn_gemm_bf16<GCN_IN><<<gemmGrid, T, 0, stream>>>(xbf, W1p, h, GCN_N);
    gcn_selfloop<<<blocks(NH), T, 0, stream>>>(h, dinv, agg, NH);
    gcn_scatter<<<scatterGrid, T, 0, stream>>>(src, dst, dinv, h, agg, GCN_E);
    gcn_zero_stats<<<1, 256, 0, stream>>>(stats);
    gcn_bn_stats<<<statsGrid, T, 0, stream>>>(agg, b1, stats, GCN_N);
    gcn_bn_finalize<<<1, 64, 0, stream>>>(g1, bt1, stats, invN);
    gcn_bn_apply<true><<<blocks(NH), T, 0, stream>>>(agg, b1, stats, nullptr, act, NH);

    // ================= layer 2 =================
    gcn_gemm_bf16<GCN_H><<<gemmGrid, T, 0, stream>>>(act, W2p, h, GCN_N);
    gcn_selfloop<<<blocks(NH), T, 0, stream>>>(h, dinv, agg, NH);
    gcn_scatter<<<scatterGrid, T, 0, stream>>>(src, dst, dinv, h, agg, GCN_E);
    gcn_zero_stats<<<1, 256, 0, stream>>>(stats);
    gcn_bn_stats<<<statsGrid, T, 0, stream>>>(agg, b2, stats, GCN_N);
    gcn_bn_finalize<<<1, 64, 0, stream>>>(g2, bt2, stats, invN);
    gcn_bn_apply<false><<<blocks(NH), T, 0, stream>>>(agg, b2, stats, out, nullptr, NH);
}
